// RelationAwareTransformer_13408887898253
// MI455X (gfx1250) — matrix operand order repaired and checked by reading, not compiled, not run on hardware
//
#include <hip/hip_runtime.h>

// Relation-aware transformer, MI455X (gfx1250), wave32 + WMMA f16->f32,
// async global->LDS double-buffered K/V staging in the fused attention kernel.
// B=4 S=1024 HID=1024 H=16 DH=64, MAX_REL=16 -> 33 distance buckets.

#define B_   4
#define S_   1024
#define HID_ 1024
#define H_   16
#define DH_  64
#define NREL 33
#define SCALE_ 0.125f   // 1/sqrt(64)

#ifndef __has_builtin
#define __has_builtin(x) 0
#endif

typedef int v4i __attribute__((ext_vector_type(4)));

#if __has_builtin(__builtin_amdgcn_global_load_async_to_lds_b128)
#define HAVE_ASYNC_LOAD 1
typedef __attribute__((address_space(1))) v4i gv4i_t;  // global int4
typedef __attribute__((address_space(3))) v4i lv4i_t;  // LDS int4
#else
#define HAVE_ASYNC_LOAD 0
#endif
#if __has_builtin(__builtin_amdgcn_s_wait_asynccnt)
#define WAIT_ASYNC(n) __builtin_amdgcn_s_wait_asynccnt(n)
#else
#define WAIT_ASYNC(n) asm volatile("s_wait_asynccnt %0" ::"n"(n) : "memory")
#endif

typedef __attribute__((ext_vector_type(16))) _Float16 v16h;
typedef __attribute__((ext_vector_type(8)))  float    v8f;

static __device__ __forceinline__ v8f wmma16(v16h a, v16h b, v8f c) {
  // (neg_a, A, neg_b, B, c_mod, C, reuse_a, reuse_b)
  return __builtin_amdgcn_wmma_f32_16x16x32_f16(false, a, false, b, (short)0, c,
                                                false, false);
}

// ---- fragment loaders -------------------------------------------------------
// A (16x32 f16): lane%16 = row M; half-wave hs holds K {hs*16..+15}.
static __device__ __forceinline__ v16h load_a_f16(const _Float16* base, int ld,
                                                  int lane) {
  int m = lane & 15, hs = lane >> 4;
  const _Float16* r = base + (size_t)m * ld + hs * 16;
  v16h a;
  for (int j = 0; j < 8; ++j) { a[j] = r[j]; a[8 + j] = r[8 + j]; }
  return a;
}
static __device__ __forceinline__ v16h load_a_f32cvt(const float* base, int ld,
                                                     int lane) {
  int m = lane & 15, hs = lane >> 4;
  const float* r = base + (size_t)m * ld + hs * 16;
  v16h a;
  for (int j = 0; j < 8; ++j) {
    a[j] = (_Float16)r[j];
    a[8 + j] = (_Float16)r[8 + j];
  }
  return a;
}
static __device__ __forceinline__ v16h load_a_lds_f32(const float* base, int ld,
                                                      int koff, int lane) {
  int m = lane & 15, hs = lane >> 4;
  const float* r = base + (size_t)m * ld + koff + hs * 16;
  v16h a;
  for (int j = 0; j < 8; ++j) {
    a[j] = (_Float16)r[j];
    a[8 + j] = (_Float16)r[8 + j];
  }
  return a;
}
// B (32x16 f16), column-contiguous storage: lane%16 = column N; lane's 16 K
// values contiguous at koff + hs*16.
static __device__ __forceinline__ v16h load_b_cols(const _Float16* base, int ld,
                                                   int koff, int lane) {
  int n = lane & 15, hs = lane >> 4;
  const _Float16* p = base + (size_t)n * ld + koff + hs * 16;
  v16h b;
  for (int j = 0; j < 16; ++j) b[j] = p[j];
  return b;
}
static __device__ __forceinline__ v16h load_b_cols_f32(const float* base, int ld,
                                                       int koff, int lane) {
  int n = lane & 15, hs = lane >> 4;
  const float* p = base + (size_t)n * ld + koff + hs * 16;
  v16h b;
  for (int j = 0; j < 16; ++j) b[j] = (_Float16)p[j];
  return b;
}
// B (32x16 f16) from a ROW-major [32][ld] LDS tile (strided column gather).
static __device__ __forceinline__ v16h load_b_rows(const _Float16* base, int ld,
                                                   int n0, int lane) {
  int n = n0 + (lane & 15), hs = lane >> 4;
  v16h b;
  for (int j = 0; j < 16; ++j) b[j] = base[(size_t)(hs * 16 + j) * ld + n];
  return b;
}

// 2KB contiguous global tile -> LDS. Async (ASYNCcnt) when available.
static __device__ __forceinline__ void stage_tile(const _Float16* g,
                                                  _Float16* l, int lane) {
#if HAVE_ASYNC_LOAD
  for (int c = 0; c < 4; ++c) {
    int off = (c * 32 + lane) * 8;  // 16B per lane per issue
    __builtin_amdgcn_global_load_async_to_lds_b128((gv4i_t*)(g + off),
                                                   (lv4i_t*)(l + off), 0, 0);
  }
#else
  for (int c = 0; c < 4; ++c) {
    int off = (c * 32 + lane) * 8;
    *(int4*)(l + off) = *(const int4*)(g + off);
  }
#endif
}

// ---- 1) per-head projections: X[b,s,:] @ W[h,:,:] + bias -> f16 [B*H,S,DH] --
__global__ __launch_bounds__(32) void proj_kernel(
    const float* __restrict__ X, const float* __restrict__ W,
    const float* __restrict__ bias, _Float16* __restrict__ out) {
  __shared__ _Float16 wT[16][40];  // W tile transposed: [e_local][k_local]
  int lane = threadIdx.x, m16 = lane & 15, hs = lane >> 4;
  int s0 = blockIdx.x * 16, e0 = blockIdx.y * 16;
  int bh = blockIdx.z, b = bh >> 4, h = bh & 15;
  const float* Xrow = X + (size_t)(b * S_ + s0) * HID_;
  const float* Wh = W + (size_t)h * HID_ * DH_;
  v8f acc; for (int i = 0; i < 8; ++i) acc[i] = 0.0f;
  for (int kk = 0; kk < HID_ / 32; ++kk) {
    int k0 = kk * 32;
    v16h a = load_a_f32cvt(Xrow + k0, HID_, lane);
    const float* wr = Wh + (size_t)(k0 + lane) * DH_ + e0;
    for (int j = 0; j < 16; ++j) wT[j][lane] = (_Float16)wr[j];
    __syncthreads();
    v16h bf = load_b_cols(&wT[0][0], 40, 0, lane);
    acc = wmma16(a, bf, acc);
    __syncthreads();
  }
  float bb = bias[h * DH_ + e0 + m16];
  _Float16* orow = out + ((size_t)bh * S_ + s0) * DH_ + e0;
  for (int i = 0; i < 8; ++i) {
    int m = i + 8 * hs;
    orow[(size_t)m * DH_ + m16] = (_Float16)(acc[i] + bb);
  }
}

// ---- 2) rb[b,h,q,d] = q_vec . rel_k_emb[d] ---------------------------------
__global__ void rbias_kernel(const _Float16* __restrict__ q16,
                             const float* __restrict__ relk,
                             float* __restrict__ rb) {
  size_t gid = (size_t)blockIdx.x * 256 + threadIdx.x;
  size_t total = (size_t)B_ * H_ * S_ * NREL;
  if (gid >= total) return;
  int d = (int)(gid % NREL);
  size_t row = gid / NREL;
  const _Float16* qv = q16 + row * DH_;
  const float* rv = relk + (size_t)d * DH_;
  float s = 0.0f;
  for (int e = 0; e < DH_; ++e) s += (float)qv[e] * rv[e];
  rb[gid] = s;
}

// ---- 3) rel_v_emb -> transposed, zero-padded f16 [64 e][64 d] ---------------
__global__ void rvT_kernel(const float* __restrict__ relv,
                           _Float16* __restrict__ rvT) {
  int t = blockIdx.x * 256 + threadIdx.x;
  if (t >= 64 * 64) return;
  int e = t >> 6, d = t & 63;
  rvT[(size_t)e * 64 + d] =
      (d < NREL) ? (_Float16)relv[(size_t)d * DH_ + e] : (_Float16)0.0f;
}

// ---- 4) fused attention, flash-style, one wave per 16 query rows ------------
__global__ __launch_bounds__(32) void attn_kernel(
    const _Float16* __restrict__ q16, const _Float16* __restrict__ k16,
    const _Float16* __restrict__ v16, const float* __restrict__ rb,
    const _Float16* __restrict__ rvT, _Float16* __restrict__ ctx) {
  __shared__ float rbs[16][NREL + 3];
  __shared__ float alphaS[16];
  __shared__ float wbins[16][64];                  // bucket mass + zero pad
  __shared__ __align__(16) _Float16 ptile[16][32]; // probs, K 16..31 zero
  __shared__ __align__(16) _Float16 ktile[2][16][64]; // K tile, double-buffered
  __shared__ __align__(16) _Float16 vtile[2][32][64]; // V row-major, k 16..31 zero
  int lane = threadIdx.x, m16 = lane & 15, hs = lane >> 4;
  int bh = blockIdx.y, q0 = blockIdx.x * 16;

  for (int idx = lane; idx < 16 * NREL; idx += 32) {
    int r = idx / NREL, d = idx - r * NREL;
    rbs[r][d] = rb[((size_t)bh * S_ + q0 + r) * NREL + d];
  }
  for (int idx = lane; idx < 16 * 64; idx += 32) wbins[idx >> 6][idx & 63] = 0.0f;
  for (int idx = lane; idx < 16 * 16; idx += 32)
    ptile[idx >> 4][16 + (idx & 15)] = (_Float16)0.0f;
  for (int idx = lane; idx < 2 * 16 * 64; idx += 32) {
    int bu = idx >> 10, rem = idx & 1023;
    vtile[bu][16 + (rem >> 6)][rem & 63] = (_Float16)0.0f;  // K-pad rows
  }
  __syncthreads();

  const _Float16* kb0 = k16 + (size_t)bh * S_ * DH_;
  const _Float16* vb0 = v16 + (size_t)bh * S_ * DH_;
  const _Float16* qbase = q16 + ((size_t)bh * S_ + q0) * DH_;
  v16h qa0 = load_a_f16(qbase, DH_, lane);
  v16h qa1 = load_a_f16(qbase + 32, DH_, lane);
  v8f acc[4];
  for (int t = 0; t < 4; ++t)
    for (int i = 0; i < 8; ++i) acc[t][i] = 0.0f;
  float mrow[8], lrow[8];
  for (int i = 0; i < 8; ++i) { mrow[i] = -1e30f; lrow[i] = 0.0f; }

#if HAVE_ASYNC_LOAD
  stage_tile(kb0, &ktile[0][0][0], lane);   // prologue: tile 0 in flight
  stage_tile(vb0, &vtile[0][0][0], lane);
#endif

  for (int kt = 0; kt < S_ / 16; ++kt) {
    int buf = kt & 1;
#if HAVE_ASYNC_LOAD
    if (kt + 1 < S_ / 16) {  // issue next tile, then wait for current (oldest 8)
      stage_tile(kb0 + (size_t)(kt + 1) * 16 * DH_, &ktile[buf ^ 1][0][0], lane);
      stage_tile(vb0 + (size_t)(kt + 1) * 16 * DH_, &vtile[buf ^ 1][0][0], lane);
      WAIT_ASYNC(8);
    } else {
      WAIT_ASYNC(0);
    }
#else
    stage_tile(kb0 + (size_t)kt * 16 * DH_, &ktile[buf][0][0], lane);
    stage_tile(vb0 + (size_t)kt * 16 * DH_, &vtile[buf][0][0], lane);
#endif
    const _Float16* kl = &ktile[buf][0][0];

    v8f s; for (int i = 0; i < 8; ++i) s[i] = 0.0f;
    s = wmma16(qa0, load_b_cols(kl, 64, 0, lane), s);
    s = wmma16(qa1, load_b_cols(kl, 64, 32, lane), s);

    float pv[8], alpha[8];
    int dst[8];
    for (int i = 0; i < 8; ++i) {
      int m = i + 8 * hs;
      int dd = (kt * 16 + m16) - (q0 + m);
      dd = dd < -16 ? -16 : (dd > 16 ? 16 : dd);
      dst[i] = dd + 16;
      float sv = (s[i] + rbs[m][dst[i]]) * SCALE_;
      float rm = sv;
      for (int off = 1; off < 16; off <<= 1) {
        float o = __shfl_xor(rm, off, 16);
        rm = rm > o ? rm : o;
      }
      float mn = mrow[i] > rm ? mrow[i] : rm;
      alpha[i] = __expf(mrow[i] - mn);
      float p = __expf(sv - mn);
      mrow[i] = mn;
      float rs = p;
      for (int off = 1; off < 16; off <<= 1) rs += __shfl_xor(rs, off, 16);
      lrow[i] = lrow[i] * alpha[i] + rs;
      pv[i] = p;
      for (int t = 0; t < 4; ++t) acc[t][i] *= alpha[i];
    }
    if (m16 == 0)
      for (int i = 0; i < 8; ++i) alphaS[i + 8 * hs] = alpha[i];
    __syncthreads();
    {  // rescale bucket masses by this step's alpha
      float a0 = alphaS[m16];
      for (int j = 0; j < 17; ++j) {
        int d = hs * 17 + j;
        if (d < NREL) wbins[m16][d] *= a0;
      }
    }
    __syncthreads();
    for (int i = 0; i < 8; ++i) {
      int m = i + 8 * hs;
      atomicAdd(&wbins[m][dst[i]], pv[i]);   // ds_add_f32
      ptile[m][m16] = (_Float16)pv[i];
    }
    __syncthreads();
    v16h pa = load_a_f16(&ptile[0][0], 32, lane);
    for (int t = 0; t < 4; ++t)
      acc[t] = wmma16(pa, load_b_rows(&vtile[buf][0][0], 64, t * 16, lane),
                      acc[t]);
    __syncthreads();
  }

  // value-bias: (16x64 bucket mass) @ (64x64 padded rel_v) via 8 WMMAs
  v16h wa0 = load_a_lds_f32(&wbins[0][0], 64, 0, lane);
  v16h wa1 = load_a_lds_f32(&wbins[0][0], 64, 32, lane);
  for (int t = 0; t < 4; ++t) {
    acc[t] = wmma16(wa0, load_b_cols(rvT + (size_t)t * 16 * 64, 64, 0, lane), acc[t]);
    acc[t] = wmma16(wa1, load_b_cols(rvT + (size_t)t * 16 * 64, 64, 32, lane), acc[t]);
  }
  int b = bh >> 4, h = bh & 15;
  for (int t = 0; t < 4; ++t)
    for (int i = 0; i < 8; ++i) {
      int m = i + 8 * hs;
      float val = acc[t][i] / lrow[i];
      ctx[(((size_t)(b * S_ + q0 + m)) * H_ + h) * DH_ + t * 16 + m16] =
          (_Float16)val;
    }
}

// ---- 5) fc: ctx[B*S,HID] @ fc_W^T + fc_b -> f32 out ------------------------
__global__ __launch_bounds__(32) void fc_kernel(
    const _Float16* __restrict__ xin, const float* __restrict__ W,
    const float* __restrict__ bias, float* __restrict__ out) {
  int lane = threadIdx.x, m16 = lane & 15, hs = lane >> 4;
  int r0 = blockIdx.x * 16, n0 = blockIdx.y * 16;
  const _Float16* xrow = xin + (size_t)r0 * HID_;
  v8f acc; for (int i = 0; i < 8; ++i) acc[i] = 0.0f;
  for (int kk = 0; kk < HID_ / 32; ++kk) {
    int k0 = kk * 32;
    v16h a = load_a_f16(xrow + k0, HID_, lane);
    v16h bf = load_b_cols_f32(W + (size_t)n0 * HID_ + k0, HID_, 0, lane);
    acc = wmma16(a, bf, acc);
  }
  float bb = bias[n0 + m16];
  for (int i = 0; i < 8; ++i) {
    int m = i + 8 * hs;
    out[(size_t)(r0 + m) * HID_ + n0 + m16] = acc[i] + bb;
  }
}

extern "C" void kernel_launch(void* const* d_in, const int* in_sizes, int n_in,
                              void* d_out, int out_size, void* d_ws,
                              size_t ws_size, hipStream_t stream) {
  (void)in_sizes; (void)n_in; (void)out_size; (void)ws_size;
  const float* query = (const float*)d_in[0];
  const float* key   = (const float*)d_in[1];
  const float* value = (const float*)d_in[2];
  const float* Wq = (const float*)d_in[3];
  const float* bq = (const float*)d_in[4];
  const float* Wk = (const float*)d_in[5];
  const float* bk = (const float*)d_in[6];
  const float* Wv = (const float*)d_in[7];
  const float* bv = (const float*)d_in[8];
  const float* relk = (const float*)d_in[9];
  const float* relv = (const float*)d_in[10];
  const float* fcW = (const float*)d_in[11];
  const float* fcb = (const float*)d_in[12];
  float* out = (float*)d_out;

  size_t nQKV = (size_t)B_ * H_ * S_ * DH_;
  _Float16* q16 = (_Float16*)d_ws;
  _Float16* k16 = q16 + nQKV;
  _Float16* v16 = k16 + nQKV;
  _Float16* ctx = v16 + nQKV;
  float* rb = (float*)(ctx + nQKV);                  // [B*H,S,33] f32
  _Float16* rvT = (_Float16*)(rb + (size_t)B_ * H_ * S_ * NREL);

  dim3 blk(32);
  dim3 gproj(S_ / 16, DH_ / 16, B_ * H_);
  proj_kernel<<<gproj, blk, 0, stream>>>(query, Wq, bq, q16);
  proj_kernel<<<gproj, blk, 0, stream>>>(key,   Wk, bk, k16);
  proj_kernel<<<gproj, blk, 0, stream>>>(value, Wv, bv, v16);

  size_t nrb = (size_t)B_ * H_ * S_ * NREL;
  rbias_kernel<<<dim3((unsigned)((nrb + 255) / 256)), dim3(256), 0, stream>>>(
      q16, relk, rb);
  rvT_kernel<<<dim3(16), dim3(256), 0, stream>>>(relv, rvT);

  dim3 gattn(S_ / 16, B_ * H_);
  attn_kernel<<<gattn, blk, 0, stream>>>(q16, k16, v16, rb, rvT, ctx);

  dim3 gfc(B_ * S_ / 16, HID_ / 16);
  fc_kernel<<<gfc, blk, 0, stream>>>(ctx, fcW, fcb, out);
}
